// ModelLSTM_71047349010535
// MI455X (gfx1250) — matrix operand order repaired and checked by reading, not compiled, not run on hardware
//
#include <hip/hip_runtime.h>

#define BB 32
#define SS 512
#define DD 512
#define HH 256
#define MMent 64
#define EE 32
#define RR 97
#define GG 1024              // 4*H (gate width per direction)
#define MTOT (BB * SS)       // 16384

typedef __attribute__((ext_vector_type(16))) __bf16          v16bf;
typedef __attribute__((ext_vector_type(8)))  float           v8f;
typedef __attribute__((ext_vector_type(8)))  unsigned short  v8us;
typedef __attribute__((ext_vector_type(16))) unsigned short  v16us;

__device__ __forceinline__ unsigned short f2bf(float f) {
  union { float f; unsigned u; } v; v.f = f;
  unsigned r = (v.u + 0x7FFFu + ((v.u >> 16) & 1u)) >> 16;
  return (unsigned short)r;
}
__device__ __forceinline__ float bf2f(unsigned short s) {
  union { unsigned u; float f; } v; v.u = ((unsigned)s) << 16;
  return v.f;
}

// A fragment 16x32 (MxK), bf16: lane<16 holds row=lr, K = {k0..k0+15};
// lane>=16 holds row=lr, K = {k0+16..k0+31}. p = &A[row*lda + k0 + half*16].
__device__ __forceinline__ v16bf frag_a(const unsigned short* p) {
  v8us lo = *(const v8us*)p;
  v8us hi = *(const v8us*)(p + 8);
  v16us c = __builtin_shufflevector(lo, hi, 0,1,2,3,4,5,6,7,8,9,10,11,12,13,14,15);
  return __builtin_bit_cast(v16bf, c);
}
// B fragment 32x16 (KxN), bf16: lane<16 holds col=lr, K = {k0..k0+15};
// lane>=16 holds col=lr, K = {k0+16..k0+31}. p = &W[col*ldb + k0 + half*16].
__device__ __forceinline__ v16bf frag_b(const unsigned short* p) {
  v8us lo = *(const v8us*)p;
  v8us hi = *(const v8us*)(p + 8);
  v16us c = __builtin_shufflevector(lo, hi, 0,1,2,3,4,5,6,7,8,9,10,11,12,13,14,15);
  return __builtin_bit_cast(v16bf, c);
}
__device__ __forceinline__ v16bf frag_zero() {
  v16us z;
#pragma unroll
  for (int i = 0; i < 16; ++i) z[i] = 0;
  return __builtin_bit_cast(v16bf, z);
}
__device__ __forceinline__ v8f wmma_bf16(v16bf a, v16bf b, v8f c) {
  return __builtin_amdgcn_wmma_f32_16x16x32_bf16(false, a, false, b, (short)0, c, false, false);
}
// Opaque zero: a value the compiler cannot prove constant.  Adding it to a pointer defeats
// LICM (prevents hoisting 512B/lane of loop-invariant weight fragments -> scratch spills)
// while preserving the pointer's global address-space provenance, so loads stay
// global_load_b128 (not flat_load, which would tie up the LDS/DScnt path).
__device__ __forceinline__ int opaque_zero() {
  int z = 0;
  asm volatile("" : "+v"(z));
  return z;
}

// ---------------- elementwise helpers ----------------

__global__ void k_cvt(const float* __restrict__ src, unsigned short* __restrict__ dst, int n) {
  int i = blockIdx.x * blockDim.x + threadIdx.x;
  if (i < n) dst[i] = f2bf(src[i]);
}

__global__ void k_add(const float* __restrict__ a, const float* __restrict__ b,
                      float* __restrict__ dst, int n) {
  int i = blockIdx.x * blockDim.x + threadIdx.x;
  if (i < n) dst[i] = a[i] + b[i];
}

__global__ void k_embed(const int* __restrict__ sents, const float* __restrict__ emb,
                        unsigned short* __restrict__ xbf, int total) {
  int i = blockIdx.x * blockDim.x + threadIdx.x;
  if (i >= total) return;
  int tok = i >> 9;                 // D = 512
  int d = i & (DD - 1);
  xbf[i] = f2bf(emb[(size_t)sents[tok] * DD + d]);
}

// ---------------- generic WMMA GEMM: C[m,n] = sum_k A[m,k]*W[n,k] + bias[n] ----------------
// block = 256 threads = 8 waves (2x4).  Block tile: 64(M) x 256(N); wave tile 32x64 (8 WMMAs/k-step).
// K multiple of 32, M multiple of 64.  RAGGED=true adds N-bound guards (only needed for N=97).
template <bool RAGGED, bool OUT_BF16>
__global__ void __launch_bounds__(256) k_gemm(
    const unsigned short* __restrict__ A,    // (M,K) bf16 row-major
    const unsigned short* __restrict__ W,    // (N,K) bf16 row-major
    const float* __restrict__ bias,          // (N)
    void* __restrict__ Cout,                 // (M,N) f32 or bf16
    int M, int N, int K, int lda, int ldb, int ldc)
{
  int lane = threadIdx.x & 31;
  int wid  = threadIdx.x >> 5;
  int wm = wid >> 2, wn = wid & 3;           // 2 x 4 wave grid
  int half = lane >> 4, lr = lane & 15;
  int m0 = blockIdx.x * 64 + wm * 32;        // wave rows [m0, m0+32)
  int n0 = blockIdx.y * 256 + wn * 64;       // wave cols [n0, n0+64)

  v8f acc[2][4];
#pragma unroll
  for (int mi = 0; mi < 2; ++mi)
#pragma unroll
    for (int j = 0; j < 4; ++j)
#pragma unroll
      for (int v = 0; v < 8; ++v) acc[mi][j][v] = 0.f;

  const unsigned short* arow0 = A + (size_t)(m0 +      lr) * lda + half * 16;
  const unsigned short* arow1 = A + (size_t)(m0 + 16 + lr) * lda + half * 16;
  for (int k0 = 0; k0 < K; k0 += 32) {
    v16bf a0 = frag_a(arow0 + k0);
    v16bf a1 = frag_a(arow1 + k0);
#pragma unroll
    for (int j = 0; j < 4; ++j) {
      int nr = n0 + j * 16 + lr;
      v16bf bf;
      if (RAGGED)
        bf = (nr < N) ? frag_b(W + (size_t)nr * ldb + k0 + half * 16) : frag_zero();
      else
        bf = frag_b(W + (size_t)nr * ldb + k0 + half * 16);
      acc[0][j] = wmma_bf16(a0, bf, acc[0][j]);
      acc[1][j] = wmma_bf16(a1, bf, acc[1][j]);
    }
  }

#pragma unroll
  for (int mi = 0; mi < 2; ++mi) {
    int mb = m0 + mi * 16 + (half ? 8 : 0);
#pragma unroll
    for (int j = 0; j < 4; ++j) {
      int n = n0 + j * 16 + lr;
      if (RAGGED && n >= N) continue;
      float bv = bias[n];
#pragma unroll
      for (int v = 0; v < 8; ++v) {
        float val = acc[mi][j][v] + bv;
        size_t idx = (size_t)(mb + v) * ldc + n;
        if constexpr (OUT_BF16) ((unsigned short*)Cout)[idx] = f2bf(val);
        else                    ((float*)Cout)[idx] = val;
      }
    }
  }
}

// ---------------- persistent WMMA LSTM recurrence (one WGP per direction) ----------------
// xp:   (B*S, 2048) bf16 precomputed input projection (+biases); dir d uses cols [d*1024, d*1024+1024)
// whh:  (2*1024, 256) bf16 recurrent weights
// hseq: (B*S, 512) bf16 output; dir d writes cols [d*256, d*256+256) at the *un-reversed* position
__global__ void __launch_bounds__(1024) k_lstm(
    const unsigned short* __restrict__ xp,
    const unsigned short* __restrict__ whh,
    unsigned short* __restrict__ hseq)
{
  extern __shared__ char smem[];
  float*          g_lds = (float*)smem;                              // 32 x 1024 f32 = 128 KB
  unsigned short* h_lds = (unsigned short*)(smem + 32 * GG * 4);     // 32 x 256 bf16 = 16 KB

  const int dir  = blockIdx.x;
  const int tid  = threadIdx.x;
  const int lane = tid & 31;
  const int wid  = tid >> 5;          // 0..31, wave owns gate cols [wid*32, wid*32+32)
  const int half = lane >> 4, lr = lane & 15;

  __builtin_amdgcn_s_wait_tensorcnt(0);   // CDNA5 split-counter path (TENSORcnt==0, free)

  // Per-wave weight rows: strip [wid*32, wid*32+32) of whh for this direction.
  const unsigned short* wrow0 = whh + (size_t)dir * GG * HH + (size_t)(wid * 32 +      lr) * HH + half * 16;
  const unsigned short* wrow1 = whh + (size_t)dir * GG * HH + (size_t)(wid * 32 + 16 + lr) * HH + half * 16;

  // init h = 0 in LDS, c = 0 in registers
#pragma unroll
  for (int u = 0; u < 8; ++u) h_lds[u * 1024 + tid] = 0;
  float c[8];
#pragma unroll
  for (int u = 0; u < 8; ++u) c[u] = 0.f;
  __syncthreads();

#pragma unroll 1
  for (int t = 0; t < SS; ++t) {
    const int s = dir ? (SS - 1 - t) : t;

    // Per-iteration opaque zero offset: keeps the weight loads inside the loop (no LICM,
    // no scratch spill) while preserving global provenance -> global_load_b128.
    const int oz = opaque_zero();
    const unsigned short* w0 = wrow0 + oz;
    const unsigned short* w1 = wrow1 + oz;

    // ---- phase 1: g = h @ whh^T  (32 x 1024, K=256), wave computes 2x2 16x16 tiles ----
    v8f acc[2][2];
#pragma unroll
    for (int mi = 0; mi < 2; ++mi)
#pragma unroll
      for (int j = 0; j < 2; ++j)
#pragma unroll
        for (int v = 0; v < 8; ++v) acc[mi][j][v] = 0.f;

#pragma unroll
    for (int k0 = 0; k0 < HH; k0 += 32) {
      v16bf a0 = frag_a(h_lds + (size_t)(0  + lr) * HH + k0 + half * 16);
      v16bf a1 = frag_a(h_lds + (size_t)(16 + lr) * HH + k0 + half * 16);
      v16bf b0 = frag_b(w0 + k0);
      acc[0][0] = wmma_bf16(a0, b0, acc[0][0]);
      acc[1][0] = wmma_bf16(a1, b0, acc[1][0]);
      v16bf b1 = frag_b(w1 + k0);
      acc[0][1] = wmma_bf16(a0, b1, acc[0][1]);
      acc[1][1] = wmma_bf16(a1, b1, acc[1][1]);
    }
#pragma unroll
    for (int mi = 0; mi < 2; ++mi)
#pragma unroll
      for (int j = 0; j < 2; ++j)
#pragma unroll
        for (int v = 0; v < 8; ++v)
          g_lds[(size_t)(mi * 16 + (half ? 8 : 0) + v) * GG + wid * 32 + j * 16 + lr] = acc[mi][j][v];
    __syncthreads();

    // ---- phase 2: gates + state update; thread owns 8 (b, j) cells across all steps ----
#pragma unroll
    for (int u = 0; u < 8; ++u) {
      int cell = u * 1024 + tid;
      int b = cell >> 8;               // 0..31
      int j = cell & 255;              // 0..255
      const unsigned short* xr = xp + ((size_t)(b * SS + s)) * (2 * GG) + (size_t)dir * GG;
      float gi = g_lds[(size_t)b * GG +        j] + bf2f(xr[j]);
      float gf = g_lds[(size_t)b * GG + 256 +  j] + bf2f(xr[256 + j]);
      float gg = g_lds[(size_t)b * GG + 512 +  j] + bf2f(xr[512 + j]);
      float go = g_lds[(size_t)b * GG + 768 +  j] + bf2f(xr[768 + j]);
      float i_ = 1.f / (1.f + __expf(-gi));
      float f_ = 1.f / (1.f + __expf(-gf));
      float o_ = 1.f / (1.f + __expf(-go));
      float cg = tanhf(gg);
      c[u] = f_ * c[u] + i_ * cg;
      float h = o_ * tanhf(c[u]);
      unsigned short hb = f2bf(h);
      h_lds[b * HH + j] = hb;
      hseq[((size_t)(b * SS + s)) * DD + dir * HH + j] = hb;
    }
    __syncthreads();
  }
}

// ---------------- span mean pooling ----------------
__global__ void __launch_bounds__(128) k_mention(
    const unsigned short* __restrict__ hseq, const int* __restrict__ ent_inds,
    float* __restrict__ mm)
{
  int bm = blockIdx.x;                 // b*64 + m
  int b = bm >> 6;
  const int* ei = ent_inds + bm * 3;
  int st = ei[1], en = ei[2];
  int len = en - st; if (len < 1) len = 1;
  float inv = 1.f / (float)len;
  for (int d = threadIdx.x; d < DD; d += 128) {
    float acc = 0.f;
    for (int s = st; s < en; ++s)
      acc += bf2f(hseq[((size_t)(b * SS + s)) * DD + d]);
    mm[(size_t)bm * DD + d] = acc * inv;
  }
}

// ---------------- entity aggregation + LayerNorm -> bf16, mask ----------------
__global__ void __launch_bounds__(256) k_entity_ln(
    const float* __restrict__ mm, const int* __restrict__ ent_inds,
    const float* __restrict__ ln_g, const float* __restrict__ ln_b,
    unsigned short* __restrict__ contbf, float* __restrict__ maskw)
{
  __shared__ float red[256];
  int be = blockIdx.x;                 // b*32 + e
  int b = be >> 5, e = be & 31;
  int t = threadIdx.x;
  float cnt = 0.f, v0 = 0.f, v1 = 0.f;
  for (int m = 0; m < MMent; ++m) {
    int id = ent_inds[(b * MMent + m) * 3];
    if (id == e) {
      cnt += 1.f;
      v0 += mm[(size_t)(b * MMent + m) * DD + t];
      v1 += mm[(size_t)(b * MMent + m) * DD + t + 256];
    }
  }
  float div = fmaxf(cnt, 1.f);
  v0 /= div; v1 /= div;
  red[t] = v0 + v1; __syncthreads();
  for (int sft = 128; sft > 0; sft >>= 1) { if (t < sft) red[t] += red[t + sft]; __syncthreads(); }
  float mu = red[0] * (1.f / 512.f); __syncthreads();
  float d0 = v0 - mu, d1 = v1 - mu;
  red[t] = d0 * d0 + d1 * d1; __syncthreads();
  for (int sft = 128; sft > 0; sft >>= 1) { if (t < sft) red[t] += red[t + sft]; __syncthreads(); }
  float rstd = rsqrtf(red[0] * (1.f / 512.f) + 1e-5f);
  contbf[(size_t)be * DD + t]       = f2bf(d0 * rstd * ln_g[t] + ln_b[t]);
  contbf[(size_t)be * DD + t + 256] = f2bf(d1 * rstd * ln_g[t + 256] + ln_b[t + 256]);
  if (t == 0) maskw[be] = fminf(cnt, 1.f);
}

// ---------------- pair sum: ps[b,e1,e2,d] = rel1[b,e2,d] + rel2[b,e1,d] (bf16) ----------------
__global__ void k_pairsum(const float* __restrict__ r1, const float* __restrict__ r2,
                          unsigned short* __restrict__ out, int total)
{
  int idx = blockIdx.x * blockDim.x + threadIdx.x;
  if (idx >= total) return;
  int d = idx & (DD - 1);
  int row = idx >> 9;
  int e2 = row & 31;
  int e1 = (row >> 5) & 31;
  int b  = row >> 10;
  out[idx] = f2bf(r1[(size_t)(b * EE + e2) * DD + d] + r2[(size_t)(b * EE + e1) * DD + d]);
}

// ---------------- mask + log_softmax over R=97 ----------------
__global__ void __launch_bounds__(128) k_logsm(const float* __restrict__ logits,
                                               const float* __restrict__ maskw,
                                               float* __restrict__ out)
{
  __shared__ float red[128];
  int r = blockIdx.x;                  // ((b*E + e1)*E + e2)
  int e2 = r & 31, e1 = (r >> 5) & 31, b = r >> 10;
  float m = maskw[b * EE + e1] * maskw[b * EE + e2];
  int t = threadIdx.x;
  float v = (t < RR) ? logits[(size_t)r * RR + t] * m : -1e30f;
  red[t] = v; __syncthreads();
  for (int sft = 64; sft > 0; sft >>= 1) { if (t < sft) red[t] = fmaxf(red[t], red[t + sft]); __syncthreads(); }
  float mx = red[0]; __syncthreads();
  float ex = (t < RR) ? __expf(v - mx) : 0.f;
  red[t] = ex; __syncthreads();
  for (int sft = 64; sft > 0; sft >>= 1) { if (t < sft) red[t] += red[t + sft]; __syncthreads(); }
  float lse = __logf(red[0]);
  if (t < RR) out[(size_t)r * RR + t] = v - mx - lse;
}

// ---------------- launcher ----------------
extern "C" void kernel_launch(void* const* d_in, const int* in_sizes, int n_in,
                              void* d_out, int out_size, void* d_ws, size_t ws_size,
                              hipStream_t stream)
{
  (void)in_sizes; (void)n_in; (void)out_size; (void)ws_size;
  const int*   sents  = (const int*)d_in[0];
  const int*   ent    = (const int*)d_in[1];
  const float* emb    = (const float*)d_in[2];
  const float* w_ih0  = (const float*)d_in[3];
  const float* w_hh0  = (const float*)d_in[4];
  const float* b_ih0  = (const float*)d_in[5];
  const float* b_hh0  = (const float*)d_in[6];
  const float* w_ih1  = (const float*)d_in[7];
  const float* w_hh1  = (const float*)d_in[8];
  const float* b_ih1  = (const float*)d_in[9];
  const float* b_hh1  = (const float*)d_in[10];
  const float* rel1_w = (const float*)d_in[11];
  const float* rel1_b = (const float*)d_in[12];
  const float* rel2_w = (const float*)d_in[13];
  const float* rel2_b = (const float*)d_in[14];
  const float* proj_w = (const float*)d_in[15];
  const float* proj_b = (const float*)d_in[16];
  const float* dec_w  = (const float*)d_in[17];
  const float* dec_b  = (const float*)d_in[18];
  const float* ln_g   = (const float*)d_in[19];
  const float* ln_b   = (const float*)d_in[20];
  float* out = (float*)d_out;

  char* ws = (char*)d_ws;
  size_t off = 0;
  auto alloc = [&](size_t bytes) -> void* {
    void* p = ws + off;
    off = (off + bytes + 255) & ~(size_t)255;
    return p;
  };

  unsigned short* wih0b  = (unsigned short*)alloc((size_t)2 * GG * DD * 2);
  unsigned short* whh0b  = (unsigned short*)alloc((size_t)2 * GG * HH * 2);
  unsigned short* wih1b  = (unsigned short*)alloc((size_t)2 * GG * DD * 2);
  unsigned short* whh1b  = (unsigned short*)alloc((size_t)2 * GG * HH * 2);
  unsigned short* r1wb   = (unsigned short*)alloc((size_t)DD * DD * 2);
  unsigned short* r2wb   = (unsigned short*)alloc((size_t)DD * DD * 2);
  unsigned short* pjwb   = (unsigned short*)alloc((size_t)DD * DD * 2);
  unsigned short* dcwb   = (unsigned short*)alloc((size_t)RR * DD * 2);
  float*          bias0  = (float*)alloc((size_t)2 * GG * 4);
  float*          bias1  = (float*)alloc((size_t)2 * GG * 4);
  unsigned short* xbf    = (unsigned short*)alloc((size_t)MTOT * DD * 2);   // emb, later hseq1
  unsigned short* hbf    = (unsigned short*)alloc((size_t)MTOT * DD * 2);   // hseq0
  unsigned short* xp     = (unsigned short*)alloc((size_t)MTOT * 2 * GG * 2);
  float*          mmw    = (float*)alloc((size_t)BB * MMent * DD * 4);
  unsigned short* contbf = (unsigned short*)alloc((size_t)BB * EE * DD * 2);
  float*          maskw  = (float*)alloc((size_t)BB * EE * 4);
  float*          rel1o  = (float*)alloc((size_t)BB * EE * DD * 4);
  float*          rel2o  = (float*)alloc((size_t)BB * EE * DD * 4);
  unsigned short* psbf   = (unsigned short*)alloc((size_t)BB * EE * EE * DD * 2);
  unsigned short* pjobf  = (unsigned short*)alloc((size_t)BB * EE * EE * DD * 2);
  float*          logits = (float*)alloc((size_t)BB * EE * EE * RR * 4);

  const int LSTM_LDS = 32 * GG * 4 + 32 * HH * 2;  // 144 KB
  (void)hipFuncSetAttribute((const void*)k_lstm,
                            hipFuncAttributeMaxDynamicSharedMemorySize, LSTM_LDS);

  // weight conversions + combined biases
  auto cvt = [&](const float* s, unsigned short* d, int n) {
    k_cvt<<<(n + 255) / 256, 256, 0, stream>>>(s, d, n);
  };
  cvt(w_ih0, wih0b, 2 * GG * DD);
  cvt(w_hh0, whh0b, 2 * GG * HH);
  cvt(w_ih1, wih1b, 2 * GG * DD);
  cvt(w_hh1, whh1b, 2 * GG * HH);
  cvt(rel1_w, r1wb, DD * DD);
  cvt(rel2_w, r2wb, DD * DD);
  cvt(proj_w, pjwb, DD * DD);
  cvt(dec_w,  dcwb, RR * DD);
  k_add<<<(2 * GG + 255) / 256, 256, 0, stream>>>(b_ih0, b_hh0, bias0, 2 * GG);
  k_add<<<(2 * GG + 255) / 256, 256, 0, stream>>>(b_ih1, b_hh1, bias1, 2 * GG);

  // embedding gather
  k_embed<<<(MTOT * DD + 255) / 256, 256, 0, stream>>>(sents, emb, xbf, MTOT * DD);

  // layer 0: input projection + recurrence
  dim3 g0(MTOT / 64, (2 * GG) / 256);              // 256 x 8
  k_gemm<false, true><<<g0, 256, 0, stream>>>(xbf, wih0b, bias0, xp,
                                              MTOT, 2 * GG, DD, DD, DD, 2 * GG);
  k_lstm<<<2, 1024, LSTM_LDS, stream>>>(xp, whh0b, hbf);

  // layer 1
  k_gemm<false, true><<<g0, 256, 0, stream>>>(hbf, wih1b, bias1, xp,
                                              MTOT, 2 * GG, DD, DD, DD, 2 * GG);
  k_lstm<<<2, 1024, LSTM_LDS, stream>>>(xp, whh1b, xbf);

  // mention / entity pooling + LayerNorm
  k_mention<<<BB * MMent, 128, 0, stream>>>(xbf, ent, mmw);
  k_entity_ln<<<BB * EE, 256, 0, stream>>>(mmw, ent, ln_g, ln_b, contbf, maskw);

  // rel1 / rel2
  dim3 gr((BB * EE) / 64, DD / 256);               // 16 x 2
  k_gemm<false, false><<<gr, 256, 0, stream>>>(contbf, r1wb, rel1_b, rel1o,
                                               BB * EE, DD, DD, DD, DD, DD);
  k_gemm<false, false><<<gr, 256, 0, stream>>>(contbf, r2wb, rel2_b, rel2o,
                                               BB * EE, DD, DD, DD, DD, DD);

  // pair sum -> proj -> dec
  int pstotal = BB * EE * EE * DD;
  k_pairsum<<<(pstotal + 255) / 256, 256, 0, stream>>>(rel1o, rel2o, psbf, pstotal);
  dim3 gp((BB * EE * EE) / 64, DD / 256);          // 512 x 2
  k_gemm<false, true><<<gp, 256, 0, stream>>>(psbf, pjwb, proj_b, pjobf,
                                              BB * EE * EE, DD, DD, DD, DD, DD);
  dim3 gd((BB * EE * EE) / 64, 1);                 // 512 x 1, ragged N=97
  k_gemm<true, false><<<gd, 256, 0, stream>>>(pjobf, dcwb, dec_b, logits,
                                              BB * EE * EE, RR, DD, DD, DD, RR);

  // masked log_softmax
  k_logsm<<<BB * EE * EE, 128, 0, stream>>>(logits, maskw, out);
}
